// BilinearSelfAttn_67310727463560
// MI455X (gfx1250) — matrix operand order repaired and checked by reading, not compiled, not run on hardware
//
#include <hip/hip_runtime.h>
#include <cstdint>
#include <cstddef>

// ---------------------------------------------------------------------------
// BiLSTM encoder + bilinear self-attention for MI455X (gfx1250, wave32, WMMA)
// B=32, T=1024, D=512, H=256 (2H==D), gates G=4H=1024
// ---------------------------------------------------------------------------

#define BB 32
#define TT 1024
#define DD 512
#define HH 256
#define GG 1024           // 4*H
#define MM (BB*TT)        // 32768

typedef __attribute__((ext_vector_type(16))) __bf16 v16bf;
typedef __attribute__((ext_vector_type(8)))  __bf16 v8bf;
typedef __attribute__((ext_vector_type(8)))  float  v8f;

// ---- WMMA fragment loaders (layouts per CDNA5 ISA 7.12.2, wave32) ----------
// A (16x32 bf16, row-major source, stride in elements):
//   lanes 0-15: m=lane, K=0..7 (V0-3) and 16..23 (V4-7)
//   lanes16-31: m=lane-16, K=8..15 and 24..31
__device__ __forceinline__ v16bf load_frag_a(const __bf16* base, int stride, int lane) {
  const int m = lane & 15, half = lane >> 4;
  const __bf16* p = base + (size_t)m * stride + half * 8;
  v8bf lo = *(const v8bf*)(p);
  v8bf hi = *(const v8bf*)(p + 16);
  v16bf a;
#pragma unroll
  for (int i = 0; i < 8; ++i) { a[i] = lo[i]; a[i + 8] = hi[i]; }
  return a;
}

// B (32x16 bf16) built from a row-major [N,K] source (we compute A * B^T):
//   lanes 0-15: n=lane, K=0..7 and 16..23 ; lanes 16-31: n=lane-16, K=8..15 and 24..31
__device__ __forceinline__ v16bf load_frag_b(const __bf16* base, int stride, int lane) {
  const int n = lane & 15, half = lane >> 4;
  const __bf16* p = base + (size_t)n * stride + half * 8;
  v8bf lo = *(const v8bf*)(p);
  v8bf hi = *(const v8bf*)(p + 16);
  v16bf b;
#pragma unroll
  for (int i = 0; i < 8; ++i) { b[i] = lo[i]; b[i + 8] = hi[i]; }
  return b;
}

__device__ __forceinline__ v8f wmma_bf16(v16bf a, v16bf b, v8f c) {
  return __builtin_amdgcn_wmma_f32_16x16x32_bf16(false, a, false, b, (short)0, c,
                                                 false, false);
}

// ---------------------------------------------------------------------------
// f32 -> bf16 cast
// ---------------------------------------------------------------------------
__global__ void cast_bf16_kernel(const float* __restrict__ in,
                                 __bf16* __restrict__ out, size_t n) {
  size_t i = (size_t)blockIdx.x * blockDim.x + threadIdx.x;
  if (i < n) out[i] = (__bf16)in[i];
}

// ---------------------------------------------------------------------------
// Generic batched WMMA GEMM:  C[M,N] = A[M,K] * B[N,K]^T (+bias[N])
// block tile 128x128, 256 threads = 8 waves (2 along M x 4 along N),
// per-wave 64x32 = 4x2 C-fragments -> 8 WMMAs per 32-wide K slab.
// Software pipelined: next slab's global loads are issued into registers
// before the current slab's LDS fragment loads / WMMAs, then committed to
// LDS after the barrier (overlaps LOADcnt latency with the WMMA pipe).
// Templated output/bias so the epilogue is branch-free straight-line code.
// All M,N,K used here are multiples of the tile sizes (no edge handling).
// ---------------------------------------------------------------------------
template <bool F32OUT, bool HAS_BIAS>
__global__ __launch_bounds__(256) void wmma_gemm_nt(
    const __bf16* __restrict__ A, size_t strideA, int lda,
    const __bf16* __restrict__ Bm, size_t strideB, int ldb,
    float* __restrict__ Cf, __bf16* __restrict__ Ch, size_t strideC, int ldc,
    int K, const float* __restrict__ bias) {
  __shared__ __bf16 As[128 * 32];
  __shared__ __bf16 Bs[128 * 32];

  const int tid = threadIdx.x;
  const int m0 = blockIdx.x * 128;
  const int n0 = blockIdx.y * 128;
  const size_t batch = blockIdx.z;

  const __bf16* Ab = A + batch * strideA;
  const __bf16* Bb = Bm + batch * strideB;

  const int wave = tid >> 5, lane = tid & 31;
  const int wm = wave >> 2, wn = wave & 3;          // 2 x 4 wave grid
  const int half = lane >> 4, nl = lane & 15;

  // staging indices: each thread moves 16 bf16 (two 16B chunks) of A and B
  const int sr = tid >> 1, sc = (tid & 1) * 16;
  const __bf16* gA = &Ab[(size_t)(m0 + sr) * lda + sc];
  const __bf16* gB = &Bb[(size_t)(n0 + sr) * ldb + sc];

  v8f acc[4][2] = {};

  // prologue: stage slab 0
  {
    *(v8bf*)&As[sr * 32 + sc]     = *(const v8bf*)(gA);
    *(v8bf*)&As[sr * 32 + sc + 8] = *(const v8bf*)(gA + 8);
    *(v8bf*)&Bs[sr * 32 + sc]     = *(const v8bf*)(gB);
    *(v8bf*)&Bs[sr * 32 + sc + 8] = *(const v8bf*)(gB + 8);
  }
  __syncthreads();

  for (int k0 = 0; k0 < K; k0 += 32) {
    const bool more = (k0 + 32) < K;

    // issue next slab's global loads (in flight across the WMMA work)
    v8bf pa0, pa1, pb0, pb1;
    if (more) {
      pa0 = *(const v8bf*)(gA + k0 + 32);
      pa1 = *(const v8bf*)(gA + k0 + 40);
      pb0 = *(const v8bf*)(gB + k0 + 32);
      pb1 = *(const v8bf*)(gB + k0 + 40);
    }

    // compute on current slab
    v16bf af[4], bf[2];
#pragma unroll
    for (int mt = 0; mt < 4; ++mt)
      af[mt] = load_frag_a(&As[(wm * 64 + mt * 16) * 32], 32, lane);
#pragma unroll
    for (int nt = 0; nt < 2; ++nt)
      bf[nt] = load_frag_b(&Bs[(wn * 32 + nt * 16) * 32], 32, lane);
#pragma unroll
    for (int mt = 0; mt < 4; ++mt)
#pragma unroll
      for (int nt = 0; nt < 2; ++nt)
        acc[mt][nt] = wmma_bf16(af[mt], bf[nt], acc[mt][nt]);

    if (more) {
      __syncthreads();  // all waves done reading the slab
      *(v8bf*)&As[sr * 32 + sc]     = pa0;
      *(v8bf*)&As[sr * 32 + sc + 8] = pa1;
      *(v8bf*)&Bs[sr * 32 + sc]     = pb0;
      *(v8bf*)&Bs[sr * 32 + sc + 8] = pb1;
      __syncthreads();
    }
  }

  // epilogue: C layout 7.12.2 (VGPR r -> M=r / M=r+8 by lane half; N=lane&15)
#pragma unroll
  for (int mt = 0; mt < 4; ++mt) {
#pragma unroll
    for (int nt = 0; nt < 2; ++nt) {
      const int col = n0 + wn * 32 + nt * 16 + nl;
      float bv = 0.0f;
      if constexpr (HAS_BIAS) bv = bias[col];
#pragma unroll
      for (int r = 0; r < 8; ++r) {
        const int row = m0 + wm * 64 + mt * 16 + half * 8 + r;
        const float v = acc[mt][nt][r] + bv;
        if constexpr (F32OUT)
          Cf[batch * strideC + (size_t)row * ldc + col] = v;
        else
          Ch[batch * strideC + (size_t)row * ldc + col] = (__bf16)v;
      }
    }
  }
}

// ---------------------------------------------------------------------------
// Persistent bidirectional LSTM recurrence. grid = 2 (dir), 512 threads
// (16 waves).  LDS (dynamic, 176KB < 320KB/WGP): gates g[32][1024] f32,
// c[32][256] f32, h[32][256] bf16.  Per step: g = h*Whh^T + xg[t] via WMMA
// (each wave owns a 64-column slice of the 1024 gate columns), then the
// pointwise LSTM cell update. Whh (0.5MB bf16) streams from L2 every step.
// NOTE: the K loop is unroll-capped to 2 — full unroll hoisted all 32 Whh
// fragment loads and spilled them to scratch (seen in round-2 asm).
// ---------------------------------------------------------------------------
__device__ __forceinline__ float sigf(float x) {
  return 1.0f / (1.0f + __expf(-x));
}

__global__ __launch_bounds__(512) void lstm_persistent_kernel(
    const float* __restrict__ xg_f, const float* __restrict__ xg_b,
    const __bf16* __restrict__ whh_f, const __bf16* __restrict__ whh_b,
    __bf16* __restrict__ xe) {
  extern __shared__ char smem_raw[];
  float*  sh_g = (float*)smem_raw;                       // 32*1024 f32 = 128KB
  float*  sh_c = (float*)(smem_raw + 32 * 1024 * 4);     // 32KB
  __bf16* sh_h = (__bf16*)(smem_raw + 32 * 1024 * 4 + 32 * 256 * 4);  // 16KB

  const int dir = blockIdx.x;
  const float*  xg  = dir ? xg_b : xg_f;   // [B,T,4H] f32
  const __bf16* whh = dir ? whh_b : whh_f; // [4H,H] bf16 row-major

  const int tid = threadIdx.x;
  const int wave = tid >> 5, lane = tid & 31;
  const int half = lane >> 4, nl = lane & 15;

  for (int i = tid; i < 32 * 256; i += 512) {
    sh_c[i] = 0.0f;
    sh_h[i] = (__bf16)0.0f;
  }
  __syncthreads();

  for (int s = 0; s < TT; ++s) {
    const int t = dir ? (TT - 1 - s) : s;

    // hint the Whh stream for this wave's column slice (global_prefetch_b8)
    __builtin_prefetch(whh + (size_t)(wave * 64) * HH, 0, 1);

    // ---- g[32 x 1024] = h[32 x 256] * Whh^T  (per wave: 64-col slice) ----
    v8f acc[2][4] = {};
#pragma unroll 2
    for (int kk = 0; kk < HH; kk += 32) {
      v16bf af[2];
#pragma unroll
      for (int mt = 0; mt < 2; ++mt)
        af[mt] = load_frag_a(&sh_h[(mt * 16) * HH + kk], HH, lane);
#pragma unroll
      for (int nt = 0; nt < 4; ++nt) {
        v16bf bf = load_frag_b(&whh[(size_t)(wave * 64 + nt * 16) * HH + kk],
                               HH, lane);
#pragma unroll
        for (int mt = 0; mt < 2; ++mt)
          acc[mt][nt] = wmma_bf16(af[mt], bf, acc[mt][nt]);
      }
    }

    // scatter accumulators + xg[t] into the shared gate buffer
#pragma unroll
    for (int mt = 0; mt < 2; ++mt) {
#pragma unroll
      for (int nt = 0; nt < 4; ++nt) {
        const int col = wave * 64 + nt * 16 + nl;
#pragma unroll
        for (int r = 0; r < 8; ++r) {
          const int b = mt * 16 + half * 8 + r;  // batch row
          sh_g[b * GG + col] =
              acc[mt][nt][r] + xg[((size_t)b * TT + t) * GG + col];
        }
      }
    }
    __syncthreads();

    // ---- pointwise cell update (gate order i,f,g,o) ----
    for (int u = tid; u < 32 * 256; u += 512) {
      const int b = u >> 8, j = u & 255;
      const float gi = sh_g[b * GG + j];
      const float gf = sh_g[b * GG + 256 + j];
      const float gg = sh_g[b * GG + 512 + j];
      const float go = sh_g[b * GG + 768 + j];
      const float c  = sigf(gf) * sh_c[u] + sigf(gi) * tanhf(gg);
      const float h  = sigf(go) * tanhf(c);
      sh_c[u] = c;
      sh_h[u] = (__bf16)h;
      xe[((size_t)b * TT + t) * DD + dir * HH + j] = (__bf16)h;
    }
    __syncthreads();
  }
}

// ---------------------------------------------------------------------------
// bf16 transpose per batch: [B,T,D] -> [B,D,T] (so A*xe fits the A*B^T GEMM)
// ---------------------------------------------------------------------------
__global__ void transpose_bf16_kernel(const __bf16* __restrict__ in,
                                      __bf16* __restrict__ out) {
  __shared__ __bf16 tile[32][33];
  const int t0 = blockIdx.x * 32, d0 = blockIdx.y * 32, b = blockIdx.z;
  for (int y = threadIdx.y; y < 32; y += 8)
    tile[y][threadIdx.x] =
        in[((size_t)b * TT + t0 + y) * DD + d0 + threadIdx.x];
  __syncthreads();
  for (int y = threadIdx.y; y < 32; y += 8)
    out[((size_t)b * DD + d0 + y) * TT + t0 + threadIdx.x] =
        tile[threadIdx.x][y];
}

// ---------------------------------------------------------------------------
// Row softmax over L[B*T, T]; query-row mask -> uniform 1/T row.
// one block per row, 256 threads x 4 elements.
// ---------------------------------------------------------------------------
__global__ __launch_bounds__(256) void softmax_kernel(
    const float* __restrict__ L, const unsigned char* __restrict__ mask,
    __bf16* __restrict__ A) {
  const size_t row = blockIdx.x;
  const float* Lr = L + row * TT;
  __bf16* Ar = A + row * TT;
  const int tid = threadIdx.x, lane = tid & 31, warp = tid >> 5;

  if (mask[row]) {  // padded query: softmax of constant row = uniform
    const __bf16 u = (__bf16)(1.0f / (float)TT);
#pragma unroll
    for (int k = 0; k < 4; ++k) Ar[tid + k * 256] = u;
    return;
  }

  __shared__ float red[8];
  float x[4], mx = -3.0e38f;
#pragma unroll
  for (int k = 0; k < 4; ++k) {
    x[k] = Lr[tid + k * 256];
    mx = fmaxf(mx, x[k]);
  }
#pragma unroll
  for (int o = 16; o > 0; o >>= 1) mx = fmaxf(mx, __shfl_xor(mx, o, 32));
  if (lane == 0) red[warp] = mx;
  __syncthreads();
  mx = red[0];
#pragma unroll
  for (int w = 1; w < 8; ++w) mx = fmaxf(mx, red[w]);
  __syncthreads();

  float sum = 0.0f;
#pragma unroll
  for (int k = 0; k < 4; ++k) sum += __expf(x[k] - mx);
#pragma unroll
  for (int o = 16; o > 0; o >>= 1) sum += __shfl_xor(sum, o, 32);
  if (lane == 0) red[warp] = sum;
  __syncthreads();
  sum = red[0];
#pragma unroll
  for (int w = 1; w < 8; ++w) sum += red[w];
  const float inv = 1.0f / sum;
#pragma unroll
  for (int k = 0; k < 4; ++k)
    Ar[tid + k * 256] = (__bf16)(__expf(x[k] - mx) * inv);
}

// ---------------------------------------------------------------------------
// host-side orchestration
// ---------------------------------------------------------------------------
extern "C" void kernel_launch(void* const* d_in, const int* in_sizes, int n_in,
                              void* d_out, int out_size, void* d_ws,
                              size_t ws_size, hipStream_t stream) {
  (void)in_sizes; (void)n_in; (void)out_size; (void)ws_size;

  const float* x       = (const float*)d_in[0];
  const unsigned char* x_mask = (const unsigned char*)d_in[1];  // bool [B,T]
  const float* W_ih_f  = (const float*)d_in[2];
  const float* W_hh_f  = (const float*)d_in[3];
  const float* b_f     = (const float*)d_in[4];
  const float* W_ih_b  = (const float*)d_in[5];
  const float* W_hh_b  = (const float*)d_in[6];
  const float* b_b     = (const float*)d_in[7];
  const float* W_l     = (const float*)d_in[8];
  float* out = (float*)d_out;

  char* ws = (char*)d_ws;
  size_t off = 0;
  auto carve = [&](size_t bytes) {
    char* p = ws + off;
    off = (off + bytes + 255) & ~(size_t)255;
    return p;
  };

  __bf16* x_h     = (__bf16*)carve((size_t)MM * DD * 2);
  __bf16* wih_f_h = (__bf16*)carve((size_t)GG * DD * 2);
  __bf16* whh_f_h = (__bf16*)carve((size_t)GG * HH * 2);
  __bf16* wih_b_h = (__bf16*)carve((size_t)GG * DD * 2);
  __bf16* whh_b_h = (__bf16*)carve((size_t)GG * HH * 2);
  __bf16* wl_h    = (__bf16*)carve((size_t)DD * DD * 2);
  float*  xg_f    = (float*) carve((size_t)MM * GG * 4);
  float*  xg_b    = (float*) carve((size_t)MM * GG * 4);
  __bf16* xe_h    = (__bf16*)carve((size_t)MM * DD * 2);
  __bf16* xeT_h   = (__bf16*)carve((size_t)MM * DD * 2);
  __bf16* proj_h  = (__bf16*)carve((size_t)MM * DD * 2);
  float*  Lbuf    = (float*) carve((size_t)BB * TT * TT * 4);
  __bf16* Abuf    = (__bf16*)carve((size_t)BB * TT * TT * 2);

  auto cast = [&](const float* src, __bf16* dst, size_t n) {
    cast_bf16_kernel<<<dim3((unsigned)((n + 255) / 256)), 256, 0, stream>>>(
        src, dst, n);
  };
  cast(x, x_h, (size_t)MM * DD);
  cast(W_ih_f, wih_f_h, (size_t)GG * DD);
  cast(W_hh_f, whh_f_h, (size_t)GG * HH);
  cast(W_ih_b, wih_b_h, (size_t)GG * DD);
  cast(W_hh_b, whh_b_h, (size_t)GG * HH);
  cast(W_l, wl_h, (size_t)DD * DD);

  // input projections: xg = x * W_ih^T + b  -> [B*T, 4H] f32
  wmma_gemm_nt<true, true><<<dim3(MM / 128, GG / 128, 1), 256, 0, stream>>>(
      x_h, 0, DD, wih_f_h, 0, DD, xg_f, nullptr, 0, GG, DD, b_f);
  wmma_gemm_nt<true, true><<<dim3(MM / 128, GG / 128, 1), 256, 0, stream>>>(
      x_h, 0, DD, wih_b_h, 0, DD, xg_b, nullptr, 0, GG, DD, b_b);

  // bidirectional recurrence -> xe bf16 [B,T,D] (fwd cols 0..255, bwd 256..511)
  const int lstm_smem = 32 * 1024 * 4 + 32 * 256 * 4 + 32 * 256 * 2;  // 176KB
  hipFuncSetAttribute((const void*)lstm_persistent_kernel,
                      hipFuncAttributeMaxDynamicSharedMemorySize, lstm_smem);
  lstm_persistent_kernel<<<2, 512, lstm_smem, stream>>>(
      xg_f, xg_b, whh_f_h, whh_b_h, xe_h);

  // proj = xe * W_l^T -> bf16 [B*T, D]
  wmma_gemm_nt<false, false><<<dim3(MM / 128, DD / 128, 1), 256, 0, stream>>>(
      xe_h, 0, DD, wl_h, 0, DD, nullptr, proj_h, 0, DD, DD, nullptr);

  // xeT for the final GEMM's B operand
  transpose_bf16_kernel<<<dim3(TT / 32, DD / 32, BB), dim3(32, 8), 0, stream>>>(
      xe_h, xeT_h);

  // scores L[b] = proj[b] * xe[b]^T  -> f32 [B,T,T]
  wmma_gemm_nt<true, false><<<dim3(TT / 128, TT / 128, BB), 256, 0, stream>>>(
      proj_h, (size_t)TT * DD, DD, xe_h, (size_t)TT * DD, DD, Lbuf, nullptr,
      (size_t)TT * TT, TT, DD, nullptr);

  // masked softmax rows -> A bf16
  softmax_kernel<<<BB * TT, 256, 0, stream>>>(Lbuf, x_mask, Abuf);

  // out[b] = A[b] * xe[b]  (== A * (xeT)^T) -> f32 [B,T,D]
  wmma_gemm_nt<true, false><<<dim3(TT / 128, DD / 128, BB), 256, 0, stream>>>(
      Abuf, (size_t)TT * TT, TT, xeT_h, (size_t)DD * TT, TT, out, nullptr,
      (size_t)TT * DD, DD, TT, nullptr);
}
